// TemporalMambaBlock_25924422598985
// MI455X (gfx1250) — matrix operand order repaired and checked by reading, not compiled, not run on hardware
//
#include <hip/hip_runtime.h>
#include <hip/hip_bf16.h>

typedef __attribute__((ext_vector_type(16))) _Float16 v16h;
typedef __attribute__((ext_vector_type(8)))  _Float16 v8h;
typedef __attribute__((ext_vector_type(4)))  _Float16 v4h;
typedef __attribute__((ext_vector_type(8)))  float    v8f;

#define T_LEN   64
#define HID     64
#define INNER   128
#define NPOS    1024

// Workspace layout (in halves): weights pre-swizzled into WMMA B-fragment order.
#define W_IN_OFF    0          // [64 x 256]  -> 16384 halves
#define W_DELTA_OFF 16384      // [128 x 128] -> 16384
#define W_SI_OFF    32768
#define W_SO_OFF    49152
#define W_OUT_OFF   65536      // [128 x 64]  -> 8192
#define WS_HALVES   73728

// ---------------------------------------------------------------------------
// Fast transcendental helpers (single v_exp/v_log/v_rcp instead of the
// correctly-rounded division/library expansions).
// ---------------------------------------------------------------------------
__device__ __forceinline__ float fast_rcp(float x)  { return __builtin_amdgcn_rcpf(x); }
__device__ __forceinline__ float fast_sigmoid(float x) {
  return fast_rcp(1.f + __expf(-x));
}
__device__ __forceinline__ float fast_tanh(float x) {
  x = fminf(10.f, fmaxf(-10.f, x));
  float e = __expf(2.f * x);
  return (e - 1.f) * fast_rcp(e + 1.f);
}
__device__ __forceinline__ float fast_softplus(float x) {
  float r = __logf(1.f + __expf(x));
  return x > 20.f ? x : r;   // folds to v_cndmask
}

// ---------------------------------------------------------------------------
// B-fragment swizzle: within each 32x16 (KxN) tile, lane-major storage, 16
// contiguous halves per lane. Per CDNA5 ISA: B fragment for f16 WMMA has
// N = lane&15, K = 8*(lane>>4) + (j&7) + 16*(j>>3) for half index j in 0..15.
// ---------------------------------------------------------------------------
__device__ __forceinline__ void frag_map(int lidx, int Nf, int& k, int& n) {
  int j    = lidx & 15;
  int lane = (lidx >> 4) & 31;
  int tile = lidx >> 9;             // 512 halves per 32x16 tile
  int ntn  = Nf >> 4;
  int nt   = tile % ntn;
  int kt   = tile / ntn;
  k = kt * 32 + ((j >> 3) << 4) + ((lane >> 4) << 3) + (j & 7);
  n = nt * 16 + (lane & 15);
}

__global__ __launch_bounds__(256)
void convert_weights_kernel(const float* __restrict__ w_in,
                            const float* __restrict__ w_delta,
                            const float* __restrict__ w_si,
                            const float* __restrict__ w_so,
                            const float* __restrict__ w_out,
                            _Float16* __restrict__ ws) {
  int idx = blockIdx.x * blockDim.x + threadIdx.x;
  if (idx >= WS_HALVES) return;
  const float* src; int Nf; int loc;
  if (idx < W_DELTA_OFF)    { src = w_in;    Nf = 256; loc = idx - W_IN_OFF;    }
  else if (idx < W_SI_OFF)  { src = w_delta; Nf = 128; loc = idx - W_DELTA_OFF; }
  else if (idx < W_SO_OFF)  { src = w_si;    Nf = 128; loc = idx - W_SI_OFF;    }
  else if (idx < W_OUT_OFF) { src = w_so;    Nf = 128; loc = idx - W_SO_OFF;    }
  else                      { src = w_out;   Nf = 64;  loc = idx - W_OUT_OFF;   }
  int k, n;
  frag_map(loc, Nf, k, n);
  ws[idx] = (_Float16)src[k * Nf + n];
}

// ---------------------------------------------------------------------------
// Fragment loaders
// ---------------------------------------------------------------------------
// A fragment (16x32 f16) from a row-major LDS tile. Per ISA layout:
// lane&15 = M row; lanes<16 get K 0..7 / 16..23, lanes>=16 get K 8..15 / 24..31.
__device__ __forceinline__ v16h ldA_lds(const _Float16* tile, int ld) {
  int lane = threadIdx.x & 31;
  const _Float16* row = tile + (lane & 15) * ld + ((lane >> 4) << 3);
  v8h lo = *(const v8h*)(row);
  v8h hi = *(const v8h*)(row + 16);
  return __builtin_shufflevector(lo, hi,
      0, 1, 2, 3, 4, 5, 6, 7, 8, 9, 10, 11, 12, 13, 14, 15);
}

// B fragment from pre-swizzled global weights: one 32B load per lane.
__device__ __forceinline__ v16h ldB_g(const _Float16* frag_tile) {
  int lane = threadIdx.x & 31;
  return *(const v16h*)(frag_tile + lane * 16);
}

// One [64x128] = signal @ [128x128] weight GEMM: 32 tiles, 4 per wave.
// Epilogue: + bias, optional tanh, f32 store to dst (row-major [64][128]).
template <bool APPLY_TANH>
__device__ __forceinline__ void gemm_sig128(const _Float16* __restrict__ sA,
                                            const _Float16* __restrict__ wB,
                                            const float* __restrict__ bias_p,
                                            float* __restrict__ dst,
                                            int wv, int lane) {
  #pragma unroll
  for (int i = 0; i < 4; ++i) {
    int idx = wv + (i << 3);          // 0..31
    int mt = idx >> 3, nt = idx & 7;
    v8f c = {};
    #pragma unroll
    for (int kk = 0; kk < 4; ++kk) {
      v16h a = ldA_lds(sA + mt * 16 * INNER + kk * 32, INNER);
      v16h b = ldB_g(wB + ((kk * 8 + nt) << 9));
      c = __builtin_amdgcn_wmma_f32_16x16x32_f16(false, a, false, b,
                                                 (short)0, c, false, false);
    }
    int n = (nt << 4) + (lane & 15);
    float bias = bias_p[n];
    int mbase = (mt << 4) + ((lane >> 4) << 3);
    #pragma unroll
    for (int r = 0; r < 8; ++r) {
      float v = c[r] + bias;
      if (APPLY_TANH) v = fast_tanh(v);
      dst[(mbase + r) * INNER + n] = v;
    }
  }
}

// ---------------------------------------------------------------------------
// Fused Mamba temporal block: one workgroup per sequence (B*N = 4096).
// ---------------------------------------------------------------------------
__global__ __launch_bounds__(256)
void mamba_block_kernel(const float* __restrict__ x,
                        const _Float16* __restrict__ ws,
                        const float* __restrict__ b_in,
                        const float* __restrict__ w_conv,
                        const float* __restrict__ b_conv,
                        const float* __restrict__ b_delta,
                        const float* __restrict__ b_si,
                        const float* __restrict__ b_so,
                        const float* __restrict__ a_log,
                        const float* __restrict__ d_skip,
                        const float* __restrict__ b_out,
                        float* __restrict__ out) {
  __shared__ _Float16 sXH[T_LEN * HID];            //  8 KB  x tile (f16)
  __shared__ _Float16 sSig[T_LEN * INNER];         // 16 KB  silu(conv(signal)) f16
  __shared__ _Float16 sGate[T_LEN * INNER];        // 16 KB  sigmoid(gate) f16
  __shared__ _Float16 sHid[T_LEN * INNER];         // 16 KB  gated scan output f16
  __shared__ float    sBig[3 * T_LEN * INNER];     // 96 KB  raw signal, then delta/si/so

  const int tid  = threadIdx.x;
  const int wv   = (tid >> 5) & 7;
  const int lane = tid & 31;
  const int s    = blockIdx.x;          // sequence index in [0, 4096)
  const int bb   = s >> 10;
  const int nn   = s & (NPOS - 1);
  const size_t base = (((size_t)bb * T_LEN) * NPOS + nn) * HID;  // + t*NPOS*HID + h

  // ---- Phase 1: stage x[seq] tile [T=64, H=64], f32 -> f16 (vectorized) ------
  for (int i = tid; i < (T_LEN * HID) / 4; i += 256) {     // 1024 float4's
    int t = i >> 4, c4 = (i & 15) << 2;
    const float4 v = *(const float4*)(x + base + (size_t)t * (NPOS * HID) + c4);
    v4h h;
    h[0] = (_Float16)v.x; h[1] = (_Float16)v.y;
    h[2] = (_Float16)v.z; h[3] = (_Float16)v.w;
    *(v4h*)(sXH + t * HID + c4) = h;
  }
  __syncthreads();

  // ---- Phase 2: GEMM1 proj = xh @ w_in + b_in ; split signal / gate ----------
  // Tile map: idx = wv + 8i, mt = idx&3, nt = idx>>2  ==>  i<4: nt in 0..7
  // (signal half, n<128), i>=4: nt in 8..15 (gate half). Branch is compile-time.
  {
    const _Float16* wB = ws + W_IN_OFF;   // [64 x 256], 16 tiles per K-slice
    #pragma unroll
    for (int i = 0; i < 8; ++i) {
      int idx = wv + (i << 3);
      int mt = idx & 3, nt = idx >> 2;
      v8f c = {};
      #pragma unroll
      for (int kk = 0; kk < 2; ++kk) {
        v16h a = ldA_lds(sXH + mt * 16 * HID + kk * 32, HID);
        v16h b = ldB_g(wB + ((kk * 16 + nt) << 9));
        c = __builtin_amdgcn_wmma_f32_16x16x32_f16(false, a, false, b,
                                                   (short)0, c, false, false);
      }
      int n = (nt << 4) + (lane & 15);
      float bias = b_in[n];
      int mbase = (mt << 4) + ((lane >> 4) << 3);
      if (i < 4) {                       // signal half: raw f32 into sBig[0]
        #pragma unroll
        for (int r = 0; r < 8; ++r)
          sBig[(mbase + r) * INNER + n] = c[r] + bias;
      } else {                           // gate half: sigmoid, f16
        int ch = n - INNER;
        #pragma unroll
        for (int r = 0; r < 8; ++r)
          sGate[(mbase + r) * INNER + ch] = (_Float16)fast_sigmoid(c[r] + bias);
      }
    }
  }
  __syncthreads();

  // ---- Phase 3: causal depthwise conv (K=4) + SiLU, per channel --------------
  if (tid < INNER) {
    const int ch = tid;
    float w0 = w_conv[ch * 4 + 0], w1 = w_conv[ch * 4 + 1];
    float w2 = w_conv[ch * 4 + 2], w3 = w_conv[ch * 4 + 3];
    float bc = b_conv[ch];
    float xm3 = 0.f, xm2 = 0.f, xm1 = 0.f;
    #pragma unroll 4
    for (int t = 0; t < T_LEN; ++t) {
      float x0 = sBig[t * INNER + ch];
      float v  = bc + w0 * xm3 + w1 * xm2 + w2 * xm1 + w3 * x0;
      sSig[t * INNER + ch] = (_Float16)(v * fast_sigmoid(v));
      xm3 = xm2; xm2 = xm1; xm1 = x0;
    }
  }
  __syncthreads();

  // ---- Phase 4: GEMMs 2-4: delta/si/so = signal @ W + b ----------------------
  gemm_sig128<false>(sSig, ws + W_DELTA_OFF, b_delta,
                     sBig + 0 * (T_LEN * INNER), wv, lane);   // delta (raw)
  gemm_sig128<true >(sSig, ws + W_SI_OFF, b_si,
                     sBig + 1 * (T_LEN * INNER), wv, lane);   // tanh(si)
  gemm_sig128<true >(sSig, ws + W_SO_OFF, b_so,
                     sBig + 2 * (T_LEN * INNER), wv, lane);   // tanh(so)
  __syncthreads();

  // ---- Phase 5: selective scan over T, per channel; apply gate ---------------
  if (tid < INNER) {
    const int ch = tid;
    float a_ch  = -fast_softplus(a_log[ch]);
    float dsk   = d_skip[ch];
    float state = 0.f;
    for (int t = 0; t < T_LEN; ++t) {
      int o = t * INNER + ch;
      float sg  = (float)sSig[o];
      float dec = __expf(fast_softplus(sBig[o]) * a_ch);
      float siv = sBig[T_LEN * INNER + o];
      float sov = sBig[2 * T_LEN * INNER + o];
      state = dec * state + siv * sg;
      float y = sov * state + dsk * sg;
      sHid[o] = (_Float16)(y * (float)sGate[o]);
    }
  }
  __syncthreads();

  // ---- Phase 6: GEMM5 out = hidden @ w_out + b_out ---------------------------
  {
    const _Float16* wB = ws + W_OUT_OFF;    // [128 x 64], 4 tiles per K-slice
    #pragma unroll
    for (int i = 0; i < 2; ++i) {
      int idx = wv + (i << 3);
      int mt = idx >> 2, nt = idx & 3;
      v8f c = {};
      #pragma unroll
      for (int kk = 0; kk < 4; ++kk) {
        v16h a = ldA_lds(sHid + mt * 16 * INNER + kk * 32, INNER);
        v16h b = ldB_g(wB + ((kk * 4 + nt) << 9));
        c = __builtin_amdgcn_wmma_f32_16x16x32_f16(false, a, false, b,
                                                   (short)0, c, false, false);
      }
      int h = (nt << 4) + (lane & 15);
      float bias = b_out[h];
      int mbase = (mt << 4) + ((lane >> 4) << 3);
      #pragma unroll
      for (int r = 0; r < 8; ++r) {
        int t = mbase + r;
        out[base + (size_t)t * (NPOS * HID) + h] = c[r] + bias;
      }
    }
  }
}

extern "C" void kernel_launch(void* const* d_in, const int* in_sizes, int n_in,
                              void* d_out, int out_size, void* d_ws, size_t ws_size,
                              hipStream_t stream) {
  const float* x       = (const float*)d_in[0];
  const float* w_in    = (const float*)d_in[1];
  const float* b_in    = (const float*)d_in[2];
  const float* w_conv  = (const float*)d_in[3];
  const float* b_conv  = (const float*)d_in[4];
  const float* w_delta = (const float*)d_in[5];
  const float* b_delta = (const float*)d_in[6];
  const float* w_si    = (const float*)d_in[7];
  const float* b_si    = (const float*)d_in[8];
  const float* w_so    = (const float*)d_in[9];
  const float* b_so    = (const float*)d_in[10];
  const float* a_log   = (const float*)d_in[11];
  const float* d_skip  = (const float*)d_in[12];
  const float* w_out   = (const float*)d_in[13];
  const float* b_out   = (const float*)d_in[14];
  _Float16* ws = (_Float16*)d_ws;

  convert_weights_kernel<<<(WS_HALVES + 255) / 256, 256, 0, stream>>>(
      w_in, w_delta, w_si, w_so, w_out, ws);

  mamba_block_kernel<<<4096, 256, 0, stream>>>(
      x, ws, b_in, w_conv, b_conv, b_delta, b_si, b_so, a_log, d_skip, b_out,
      (float*)d_out);
}
